// gPool_5617817223700
// MI455X (gfx1250) — matrix operand order repaired and checked by reading, not compiled, not run on hardware
//
#include <hip/hip_runtime.h>
#include <hip/hip_bf16.h>
#include <stdint.h>

#define HS   64
#define FF   128
#define NH   4
#define DH   16
#define NTOK 16384
#define KK   4096
#define BB   2
#define EPSV 1e-5f
#define TILE 2048

typedef __attribute__((ext_vector_type(16))) _Float16 v16h;
typedef __attribute__((ext_vector_type(8)))  float    v8f;

union AF { v16h v; unsigned d[8]; };

__device__ __forceinline__ v8f wmma16(v16h a, v16h b, v8f c) {
  return __builtin_amdgcn_wmma_f32_16x16x32_f16(false, a, false, b, (short)0, c, false, false);
}

// ---- fragment loaders (CDNA5 ISA 7.12.2 wave32 layouts) ----
// A 16x32 f16: lane l holds row (l&15); lane<16: K pairs {0,2,4,6},{16,18,20,22}
//                                       lane>=16: {8,10,12,14},{24,26,28,30}
__device__ __forceinline__ v16h load_a16(const _Float16* A, int ld, int row0, int k0) {
  int l = threadIdx.x & 31;
  const unsigned* p = (const unsigned*)(A + (size_t)(row0 + (l & 15)) * ld);
  int koff = k0 + ((l & 16) ? 8 : 0);
  AF f;
#pragma unroll
  for (int i = 0; i < 4; ++i) f.d[i]     = p[(koff >> 1) + i];
#pragma unroll
  for (int i = 0; i < 4; ++i) f.d[4 + i] = p[((koff + 16) >> 1) + i];
  return f.v;
}
__device__ __forceinline__ v16h load_a16_pad(const _Float16* A, int ld, int row0, int kvalid) {
  int l = threadIdx.x & 31;
  const unsigned* p = (const unsigned*)(A + (size_t)(row0 + (l & 15)) * ld);
  int koff = (l & 16) ? 8 : 0;
  AF f;
#pragma unroll
  for (int i = 0; i < 4; ++i) { int k = koff + 2 * i;      f.d[i]     = (k < kvalid) ? p[k >> 1] : 0u; }
#pragma unroll
  for (int i = 0; i < 4; ++i) { int k = 16 + koff + 2 * i; f.d[4 + i] = (k < kvalid) ? p[k >> 1] : 0u; }
  return f.v;
}
// B 32x16 f16 from Bt stored [col][k]: lane = col (l&15); lanes 0-15 K k0+{0..7,16..23}, lanes 16-31 K k0+{8..15,24..31}
__device__ __forceinline__ v16h load_b16(const _Float16* Bt, int ldk, int col0, int k0) {
  int l = threadIdx.x & 31;
  int kbase = k0 + ((l & 16) ? 8 : 0);
  const unsigned* p = (const unsigned*)(Bt + (size_t)(col0 + (l & 15)) * ldk);
  AF f;
#pragma unroll
  for (int i = 0; i < 8; ++i) f.d[i] = p[((kbase + ((i & 4) ? 16 : 0)) >> 1) + (i & 3)];
  return f.v;
}
__device__ __forceinline__ v16h load_b16_pad(const _Float16* Bt, int ldk, int col0, int kvalid) {
  int l = threadIdx.x & 31;
  int kbase = (l & 16) ? 8 : 0;
  const unsigned* p = (const unsigned*)(Bt + (size_t)(col0 + (l & 15)) * ldk);
  AF f;
#pragma unroll
  for (int i = 0; i < 8; ++i) { int k = kbase + 2 * (i & 3) + ((i & 4) ? 16 : 0); f.d[i] = (k < kvalid) ? p[k >> 1] : 0u; }
  return f.v;
}

// ================= stage kernels =================
__global__ void k_score(const float* __restrict__ x, const float* __restrict__ p,
                        float* __restrict__ y) {
  int t = blockIdx.x * blockDim.x + threadIdx.x;
  if (t >= BB * NTOK) return;
  float pn = 0.f;
#pragma unroll
  for (int i = 0; i < HS; ++i) pn += p[i] * p[i];
  float inv = rsqrtf(pn);
  const float* xr = x + (size_t)t * HS;
  float s = 0.f;
#pragma unroll
  for (int i = 0; i < HS; ++i) s += xr[i] * p[i];
  y[t] = s * inv;
}

__global__ void k_sort_init(const float* __restrict__ y, unsigned long long* __restrict__ keys) {
  int t = blockIdx.x * blockDim.x + threadIdx.x;
  if (t >= NTOK) return;
  unsigned u = __float_as_uint(y[t]);             // batch-0 scores
  u = (u & 0x80000000u) ? ~u : (u | 0x80000000u);  // order-preserving map
  keys[t] = ((unsigned long long)u << 32) | (0xFFFFFFFFu - (unsigned)t); // ties: low idx first
}

// Fused bitonic stages inside a 2048-element LDS tile.
// kstart==2,kend==TILE: full local sort.  kstart==kend==k>TILE: merge stages j<=TILE/2.
__global__ void k_sort_local(unsigned long long* __restrict__ keys, int kstart, int kend) {
  __shared__ unsigned long long s[TILE];
  int t = threadIdx.x;                 // 1024 threads
  int base = blockIdx.x * TILE;
  s[t] = keys[base + t];
  s[t + 1024] = keys[base + t + 1024];
  __syncthreads();
  for (int k = kstart; k <= kend; k <<= 1) {
    for (int j = (k > TILE ? TILE / 2 : k >> 1); j > 0; j >>= 1) {
      int i1 = ((t & ~(j - 1)) << 1) | (t & (j - 1));
      int i2 = i1 + j;
      bool up = (((base + i1) & k) == 0);   // descending network
      unsigned long long a = s[i1], b = s[i2];
      if (up ? (a < b) : (a > b)) { s[i1] = b; s[i2] = a; }
      __syncthreads();
    }
  }
  keys[base + t] = s[t];
  keys[base + t + 1024] = s[t + 1024];
}

// Global compare-exchange pass for j >= TILE (cross-tile stages only).
__global__ void k_sort_pass(unsigned long long* __restrict__ keys, int kk_, int jj_) {
  int i = blockIdx.x * blockDim.x + threadIdx.x;
  if (i >= NTOK) return;
  int ixj = i ^ jj_;
  if (ixj <= i) return;
  unsigned long long a = keys[i], b = keys[ixj];
  bool up = (i & kk_) == 0;
  if (up ? (a < b) : (a > b)) { keys[i] = b; keys[ixj] = a; }
}

__global__ void k_extract(const unsigned long long* __restrict__ keys,
                          int* __restrict__ idx, float* __restrict__ dout) {
  int t = blockIdx.x * blockDim.x + threadIdx.x;
  if (t >= KK) return;
  unsigned id = 0xFFFFFFFFu - (unsigned)(keys[t] & 0xFFFFFFFFu);
  idx[t] = (int)id;
  dout[(size_t)BB * KK * HS + 1 + t] = (float)id;
  if (t == 0) dout[(size_t)BB * KK * HS] = (float)NTOK;
}

// Wt[n*K + k] = W[k*N + n]
__global__ void k_w2h(const float* __restrict__ W, _Float16* __restrict__ Wt, int K, int Nn) {
  int t = blockIdx.x * blockDim.x + threadIdx.x;
  if (t >= K * Nn) return;
  int n = t / K, k = t % K;
  Wt[t] = (_Float16)W[(size_t)k * Nn + n];
}

__global__ void k_gather(const float* __restrict__ x, const float* __restrict__ y,
                         const int* __restrict__ idx, const float* __restrict__ g0,
                         const float* __restrict__ b0, float* __restrict__ h32,
                         _Float16* __restrict__ h16) {
  int t = blockIdx.x * blockDim.x + threadIdx.x;
  if (t >= BB * KK) return;
  int b = t / KK, r = t % KK;
  int src = idx[r];
  float gate = 1.f / (1.f + __expf(-y[(size_t)b * NTOK + src]));
  const float* xr = x + ((size_t)b * NTOK + src) * HS;
  float mean = 0.f, sq = 0.f;
#pragma unroll
  for (int i = 0; i < HS; ++i) { float v = xr[i] * gate; mean += v; sq += v * v; }
  mean *= (1.f / HS);
  float inv = rsqrtf(sq * (1.f / HS) - mean * mean + EPSV);
  float* o32 = h32 + (size_t)t * HS;
  _Float16* o16 = h16 + (size_t)t * HS;
#pragma unroll
  for (int i = 0; i < HS; ++i) {
    float hv = (xr[i] * gate - mean) * inv * g0[i] + b0[i];
    o32[i] = hv; o16[i] = (_Float16)hv;
  }
}

__global__ void k_ln(const float* __restrict__ in, const float* __restrict__ g,
                     const float* __restrict__ bb, float* __restrict__ out32,
                     _Float16* __restrict__ out16) {
  int t = blockIdx.x * blockDim.x + threadIdx.x;
  if (t >= BB * KK) return;
  const float* xr = in + (size_t)t * HS;
  float mean = 0.f, sq = 0.f;
#pragma unroll
  for (int i = 0; i < HS; ++i) { float v = xr[i]; mean += v; sq += v * v; }
  mean *= (1.f / HS);
  float inv = rsqrtf(sq * (1.f / HS) - mean * mean + EPSV);
#pragma unroll
  for (int i = 0; i < HS; ++i) {
    float hv = (xr[i] - mean) * inv * g[i] + bb[i];
    out32[(size_t)t * HS + i] = hv;
    if (out16) out16[(size_t)t * HS + i] = (_Float16)hv;
  }
}

// Q/K/V projection: block = 4 waves (one head/ntile each), one 16-row tile per block.
__global__ void k_qkv(const _Float16* __restrict__ h16, const _Float16* __restrict__ wqT,
                      const _Float16* __restrict__ wkT, const _Float16* __restrict__ wvT,
                      const float* __restrict__ bq, const float* __restrict__ bk,
                      const float* __restrict__ bv, _Float16* __restrict__ q16,
                      _Float16* __restrict__ k16, _Float16* __restrict__ vT) {
  int blk = blockIdx.x;
  int b = blk / (KK / 16), mt = blk % (KK / 16);
  int w = threadIdx.x >> 5, lane = threadIdx.x & 31;
  const _Float16* A = h16 + (size_t)b * KK * HS;
  v16h a0 = load_a16(A, HS, mt * 16, 0);
  v16h a1 = load_a16(A, HS, mt * 16, 32);
  int d = lane & 15, hi = (lane & 16) ? 8 : 0, col = w * 16 + d;
  const _Float16* Wts[3] = { wqT, wkT, wvT };
  const float* Bs[3] = { bq, bk, bv };
#pragma unroll
  for (int m = 0; m < 3; ++m) {
    v16h b0 = load_b16(Wts[m], HS, w * 16, 0);
    v16h b1 = load_b16(Wts[m], HS, w * 16, 32);
    v8f c = {};
    c = wmma16(a0, b0, c);
    c = wmma16(a1, b1, c);
#pragma unroll
    for (int r = 0; r < 8; ++r) {
      int row = mt * 16 + r + hi;
      float val = c[r] + Bs[m][col];
      if (m == 0)      q16[(((size_t)b * NH + w) * KK + row) * DH + d] = (_Float16)(val * 0.25f);
      else if (m == 1) k16[(((size_t)b * NH + w) * KK + row) * DH + d] = (_Float16)val;
      else             vT [(((size_t)b * NH + w) * DH + d) * KK + row] = (_Float16)val;
    }
  }
}

// Flash attention, split-K: block = 4 waves cooperating on one (b,head,16-row) tile;
// each wave covers 1024 keys, partials merged in LDS (8 waves/SIMD occupancy).
__global__ void k_attn(const _Float16* __restrict__ q16, const _Float16* __restrict__ k16,
                       const _Float16* __restrict__ vT, _Float16* __restrict__ o16) {
  __shared__ __align__(16) _Float16 pb[4][16 * 32];
  __shared__ float accS[4][16][16];
  __shared__ float mS[4][16], lS[4][16];
  int w = threadIdx.x >> 5, lane = threadIdx.x & 31;
  int task = blockIdx.x;                       // [0, BB*NH*256)
  int b = task / (NH * (KK / 16));
  int rem = task % (NH * (KK / 16));
  int h = rem / (KK / 16), mt = rem % (KK / 16);
  const _Float16* Q  = q16 + ((size_t)b * NH + h) * KK * DH;
  const _Float16* Kp = k16 + ((size_t)b * NH + h) * KK * DH;
  const _Float16* Vt = vT  + ((size_t)b * NH + h) * DH * KK;
  v16h aq = load_a16_pad(Q, DH, mt * 16, DH);
  v8f acc = {};
  float mrow[8], lrow[8];
#pragma unroll
  for (int r = 0; r < 8; ++r) { mrow[r] = -3.0e38f; lrow[r] = 0.f; }
  int c = lane & 15, hi = (lane & 16) ? 8 : 0;
  _Float16* pw = &pb[w][0];
  int kt0 = w * (KK / 32 / 4), kt1 = kt0 + (KK / 32 / 4);
  for (int kt = kt0; kt < kt1; ++kt) {
    int kb = kt * 32;
    __builtin_prefetch(Kp + (size_t)(kb + 32) * DH, 0, 0);
    v16h bk0 = load_b16_pad(Kp, DH, kb, DH);
    v16h bk1 = load_b16_pad(Kp, DH, kb + 16, DH);
    v8f z = {};
    v8f s0 = wmma16(aq, bk0, z);
    v8f s1 = wmma16(aq, bk1, z);
#pragma unroll
    for (int r = 0; r < 8; ++r) {
      float mx = fmaxf(s0[r], s1[r]);
      mx = fmaxf(mx, __shfl_xor(mx, 1, 16));
      mx = fmaxf(mx, __shfl_xor(mx, 2, 16));
      mx = fmaxf(mx, __shfl_xor(mx, 4, 16));
      mx = fmaxf(mx, __shfl_xor(mx, 8, 16));
      float mn = fmaxf(mrow[r], mx);
      float corr = __expf(mrow[r] - mn);
      mrow[r] = mn;
      float p0 = __expf(s0[r] - mn);
      float p1 = __expf(s1[r] - mn);
      float rs = p0 + p1;
      rs += __shfl_xor(rs, 1, 16);
      rs += __shfl_xor(rs, 2, 16);
      rs += __shfl_xor(rs, 4, 16);
      rs += __shfl_xor(rs, 8, 16);
      lrow[r] = lrow[r] * corr + rs;
      acc[r] *= corr;
      pw[(r + hi) * 32 + c]      = (_Float16)p0;
      pw[(r + hi) * 32 + c + 16] = (_Float16)p1;
    }
    asm volatile("s_wait_dscnt 0x0" ::: "memory");  // cross-lane LDS RAW before A-frag read
    v16h ap = load_a16(pw, 32, 0, 0);
    v16h bv = load_b16(Vt, KK, 0, kb);
    acc = wmma16(ap, bv, acc);
  }
  // publish per-wave partials
#pragma unroll
  for (int r = 0; r < 8; ++r) accS[w][r + hi][c] = acc[r];
  if (c == 0) {
#pragma unroll
    for (int r = 0; r < 8; ++r) { mS[w][r + hi] = mrow[r]; lS[w][r + hi] = lrow[r]; }
  }
  __syncthreads();
  // merge 4 partials: 256 output elems over 128 threads
  for (int e = threadIdx.x; e < 256; e += 128) {
    int row = e >> 4, col = e & 15;
    float m = fmaxf(fmaxf(mS[0][row], mS[1][row]), fmaxf(mS[2][row], mS[3][row]));
    float l = 0.f, o = 0.f;
#pragma unroll
    for (int w4 = 0; w4 < 4; ++w4) {
      float f = __expf(mS[w4][row] - m);
      l += lS[w4][row] * f;
      o += accS[w4][row][col] * f;
    }
    o16[((size_t)b * KK + mt * 16 + row) * HS + h * DH + col] = (_Float16)(o / l);
  }
}

// t = o @ wo + bo + h   (f32)
__global__ void k_ogemm(const _Float16* __restrict__ o16, const _Float16* __restrict__ woT,
                        const float* __restrict__ bo, const float* __restrict__ h32,
                        float* __restrict__ t) {
  int blk = blockIdx.x;
  int b = blk / (KK / 16), mt = blk % (KK / 16);
  int w = threadIdx.x >> 5, lane = threadIdx.x & 31;
  const _Float16* A = o16 + (size_t)b * KK * HS;
  v16h a0 = load_a16(A, HS, mt * 16, 0);
  v16h a1 = load_a16(A, HS, mt * 16, 32);
  v16h b0 = load_b16(woT, HS, w * 16, 0);
  v16h b1 = load_b16(woT, HS, w * 16, 32);
  v8f c = {};
  c = wmma16(a0, b0, c);
  c = wmma16(a1, b1, c);
  int n = w * 16 + (lane & 15), hi = (lane & 16) ? 8 : 0;
#pragma unroll
  for (int r = 0; r < 8; ++r) {
    size_t o = ((size_t)b * KK + mt * 16 + r + hi) * HS + n;
    t[o] = c[r] + bo[n] + h32[o];
  }
}

// ff1 = relu(h2 @ w1 + b1) f16; block = 8 waves (128 cols)
__global__ void k_ff1(const _Float16* __restrict__ h2, const _Float16* __restrict__ w1T,
                      const float* __restrict__ b1, _Float16* __restrict__ ff1) {
  int blk = blockIdx.x;
  int b = blk / (KK / 16), mt = blk % (KK / 16);
  int w = threadIdx.x >> 5, lane = threadIdx.x & 31;
  const _Float16* A = h2 + (size_t)b * KK * HS;
  v16h a0 = load_a16(A, HS, mt * 16, 0);
  v16h a1 = load_a16(A, HS, mt * 16, 32);
  v16h b0 = load_b16(w1T, HS, w * 16, 0);
  v16h b1f = load_b16(w1T, HS, w * 16, 32);
  v8f c = {};
  c = wmma16(a0, b0, c);
  c = wmma16(a1, b1f, c);
  int n = w * 16 + (lane & 15), hi = (lane & 16) ? 8 : 0;
#pragma unroll
  for (int r = 0; r < 8; ++r) {
    float v = c[r] + b1[n];
    ff1[((size_t)b * KK + mt * 16 + r + hi) * FF + n] = (_Float16)fmaxf(v, 0.f);
  }
}

// t2 = ff1 @ w2 + b2 + h2 (f32)
__global__ void k_ff2(const _Float16* __restrict__ ff1, const _Float16* __restrict__ w2T,
                      const float* __restrict__ b2, const float* __restrict__ h2_32,
                      float* __restrict__ t2) {
  int blk = blockIdx.x;
  int b = blk / (KK / 16), mt = blk % (KK / 16);
  int w = threadIdx.x >> 5, lane = threadIdx.x & 31;
  const _Float16* A = ff1 + (size_t)b * KK * FF;
  v8f c = {};
#pragma unroll
  for (int i = 0; i < 4; ++i) {
    v16h a = load_a16(A, FF, mt * 16, 32 * i);
    v16h bf = load_b16(w2T, FF, w * 16, 32 * i);
    c = wmma16(a, bf, c);
  }
  int n = w * 16 + (lane & 15), hi = (lane & 16) ? 8 : 0;
#pragma unroll
  for (int r = 0; r < 8; ++r) {
    size_t o = ((size_t)b * KK + mt * 16 + r + hi) * HS + n;
    t2[o] = c[r] + b2[n] + h2_32[o];
  }
}

// ================= host side =================
extern "C" void kernel_launch(void* const* d_in, const int* in_sizes, int n_in,
                              void* d_out, int out_size, void* d_ws, size_t ws_size,
                              hipStream_t stream) {
  const float* x     = (const float*)d_in[0];
  const float* p     = (const float*)d_in[1];
  const float* ln0_g = (const float*)d_in[2];
  const float* ln0_b = (const float*)d_in[3];
  const float* wq    = (const float*)d_in[4];
  const float* bq    = (const float*)d_in[5];
  const float* wk    = (const float*)d_in[6];
  const float* bk    = (const float*)d_in[7];
  const float* wv    = (const float*)d_in[8];
  const float* bv    = (const float*)d_in[9];
  const float* wo    = (const float*)d_in[10];
  const float* bo    = (const float*)d_in[11];
  const float* ln1_g = (const float*)d_in[12];
  const float* ln1_b = (const float*)d_in[13];
  const float* w1    = (const float*)d_in[14];
  const float* b1    = (const float*)d_in[15];
  const float* w2    = (const float*)d_in[16];
  const float* b2    = (const float*)d_in[17];
  const float* ln2_g = (const float*)d_in[18];
  const float* ln2_b = (const float*)d_in[19];
  float* out = (float*)d_out;
  char* ws = (char*)d_ws;

  size_t off = 0;
  auto carve = [&](size_t bytes) { char* r = ws + off; off += (bytes + 255) & ~(size_t)255; return r; };
  float*              y     = (float*)carve((size_t)BB * NTOK * 4);
  unsigned long long* keys  = (unsigned long long*)carve((size_t)NTOK * 8);
  int*                idx   = (int*)carve((size_t)KK * 4);
  float*              h32   = (float*)carve((size_t)BB * KK * HS * 4);
  _Float16*           h16   = (_Float16*)carve((size_t)BB * KK * HS * 2);
  _Float16*           q16   = (_Float16*)carve((size_t)BB * KK * HS * 2);
  _Float16*           k16   = (_Float16*)carve((size_t)BB * KK * HS * 2);
  _Float16*           vT    = (_Float16*)carve((size_t)BB * KK * HS * 2);
  _Float16*           o16   = (_Float16*)carve((size_t)BB * KK * HS * 2);
  float*              tbuf  = (float*)carve((size_t)BB * KK * HS * 4);
  float*              h2_32 = (float*)carve((size_t)BB * KK * HS * 4);
  _Float16*           h2_16 = (_Float16*)carve((size_t)BB * KK * HS * 2);
  _Float16*           ff1   = (_Float16*)carve((size_t)BB * KK * FF * 2);
  float*              t2    = (float*)carve((size_t)BB * KK * HS * 4);
  _Float16*           wqT   = (_Float16*)carve((size_t)HS * HS * 2);
  _Float16*           wkT   = (_Float16*)carve((size_t)HS * HS * 2);
  _Float16*           wvT   = (_Float16*)carve((size_t)HS * HS * 2);
  _Float16*           woT   = (_Float16*)carve((size_t)HS * HS * 2);
  _Float16*           w1T   = (_Float16*)carve((size_t)HS * FF * 2);
  _Float16*           w2T   = (_Float16*)carve((size_t)FF * HS * 2);

  k_score<<<(BB * NTOK) / 256, 256, 0, stream>>>(x, p, y);
  k_sort_init<<<NTOK / 256, 256, 0, stream>>>(y, keys);
  // fused bitonic: 10 launches instead of 105
  k_sort_local<<<NTOK / TILE, 1024, 0, stream>>>(keys, 2, TILE);
  for (int k = TILE * 2; k <= NTOK; k <<= 1) {
    for (int j = k >> 1; j >= TILE; j >>= 1)
      k_sort_pass<<<NTOK / 256, 256, 0, stream>>>(keys, k, j);
    k_sort_local<<<NTOK / TILE, 1024, 0, stream>>>(keys, k, k);
  }
  k_extract<<<KK / 256, 256, 0, stream>>>(keys, idx, out);

  k_w2h<<<(HS * HS + 255) / 256, 256, 0, stream>>>(wq, wqT, HS, HS);
  k_w2h<<<(HS * HS + 255) / 256, 256, 0, stream>>>(wk, wkT, HS, HS);
  k_w2h<<<(HS * HS + 255) / 256, 256, 0, stream>>>(wv, wvT, HS, HS);
  k_w2h<<<(HS * HS + 255) / 256, 256, 0, stream>>>(wo, woT, HS, HS);
  k_w2h<<<(HS * FF + 255) / 256, 256, 0, stream>>>(w1, w1T, HS, FF);
  k_w2h<<<(FF * HS + 255) / 256, 256, 0, stream>>>(w2, w2T, FF, HS);

  k_gather<<<(BB * KK) / 256, 256, 0, stream>>>(x, y, idx, ln0_g, ln0_b, h32, h16);
  k_qkv<<<BB * (KK / 16), 128, 0, stream>>>(h16, wqT, wkT, wvT, bq, bk, bv, q16, k16, vT);
  k_attn<<<BB * NH * (KK / 16), 128, 0, stream>>>(q16, k16, vT, o16);
  k_ogemm<<<BB * (KK / 16), 128, 0, stream>>>(o16, woT, bo, h32, tbuf);
  k_ln<<<(BB * KK) / 256, 256, 0, stream>>>(tbuf, ln1_g, ln1_b, h2_32, h2_16);
  k_ff1<<<BB * (KK / 16), 256, 0, stream>>>(h2_16, w1T, b1, ff1);
  k_ff2<<<BB * (KK / 16), 128, 0, stream>>>(ff1, w2T, b2, h2_32, t2);
  k_ln<<<(BB * KK) / 256, 256, 0, stream>>>(t2, ln2_g, ln2_b, out, nullptr);
}
